// DSnetwork_28432683499908
// MI455X (gfx1250) — matrix operand order repaired and checked by reading, not compiled, not run on hardware
//
#include <hip/hip_runtime.h>
#include <hip/hip_bf16.h>

// Problem constants (fixed in the reference module).
#define N_ 500000
#define D_ 128
#define G_ 10000
#define T_ 10

typedef __attribute__((ext_vector_type(16))) __bf16 bf16x16;
typedef __attribute__((ext_vector_type(8)))  float  f32x8;

union Frag {
  bf16x16 v;
  unsigned u[8];
  uint4 q[2];
};

// RNE fp32 -> bf16 (returns 16-bit pattern), and the residual float.
__device__ __forceinline__ unsigned bf16_rne(float x) {
  unsigned u = __float_as_uint(x);
  return (u + 0x7FFFu + ((u >> 16) & 1u)) >> 16;
}
__device__ __forceinline__ void split_bf16(float x, unsigned& hi, unsigned& lo) {
  hi = bf16_rne(x);
  float r = x - __uint_as_float(hi << 16);
  lo = bf16_rne(r);
}

// ---------------------------------------------------------------------------
// Pack W (row-major [k][j]) into two transposed bf16 planes, dword = bf16 pair
// of consecutive K (even K in [15:0]):  WtHi/WtLo[j*64 + k/2].
// ---------------------------------------------------------------------------
__global__ void k_packW(const float* __restrict__ W,
                        unsigned* __restrict__ WtHi, unsigned* __restrict__ WtLo) {
  int i = blockIdx.x * blockDim.x + threadIdx.x;   // 0 .. 128*64-1
  if (i >= D_ * (D_ / 2)) return;
  int j = i >> 6;          // output column
  int kd = i & 63;         // K dword (covers K=2kd, 2kd+1)
  float x0 = W[(2 * kd + 0) * D_ + j];
  float x1 = W[(2 * kd + 1) * D_ + j];
  unsigned h0, l0, h1, l1;
  split_bf16(x0, h0, l0);
  split_bf16(x1, h1, l1);
  WtHi[j * (D_ / 2) + kd] = h0 | (h1 << 16);
  WtLo[j * (D_ / 2) + kd] = l0 | (l1 << 16);
}

// ---------------------------------------------------------------------------
// Deterministic segment-mean: one block per graph; batch_idx is sorted, so
// binary-search the contiguous row range [s,e) and sum sequentially.
// Stores the MEAN directly (1/max(count,1) folded in).
// ---------------------------------------------------------------------------
__global__ __launch_bounds__(128) void k_pool(const float* __restrict__ h,
                                              const int* __restrict__ bidx,
                                              float* __restrict__ pooled, int n) {
  int g = blockIdx.x;
  int lo = 0, hi = n;
  while (lo < hi) { int m = (lo + hi) >> 1; if (bidx[m] < g) lo = m + 1; else hi = m; }
  int s = lo;
  hi = n;
  while (lo < hi) { int m = (lo + hi) >> 1; if (bidx[m] < g + 1) lo = m + 1; else hi = m; }
  int e = lo;
  int tid = threadIdx.x;
  float acc = 0.f;
  for (int r = s; r < e; ++r) acc += h[(size_t)r * D_ + tid];
  float invc = 1.0f / fmaxf((float)(e - s), 1.0f);
  pooled[(size_t)g * D_ + tid] = acc * invc;
}

// ---------------------------------------------------------------------------
// x2[g][j] = sum_k pooled[g][k]*Ws[k][j] + bs[j] + bmain[j]   (8 graphs/block)
// ---------------------------------------------------------------------------
__global__ __launch_bounds__(128) void k_x2(const float* __restrict__ pooled,
                                            const float* __restrict__ Ws,
                                            const float* __restrict__ bs,
                                            const float* __restrict__ bmain,
                                            float* __restrict__ x2) {
  __shared__ float sp[8][D_];
  int g0 = blockIdx.x * 8;
  int tid = threadIdx.x;
  for (int q = 0; q < 8; ++q) {
    int g = g0 + q;
    sp[q][tid] = (g < G_) ? pooled[(size_t)g * D_ + tid] : 0.f;
  }
  __syncthreads();
  float acc[8];
#pragma unroll
  for (int q = 0; q < 8; ++q) acc[q] = 0.f;
  for (int k = 0; k < D_; ++k) {
    float w = Ws[k * D_ + tid];
#pragma unroll
    for (int q = 0; q < 8; ++q) acc[q] += sp[q][k] * w;
  }
  float bb = bs[tid] + bmain[tid];
  for (int q = 0; q < 8; ++q) {
    int g = g0 + q;
    if (g < G_) x2[(size_t)g * D_ + tid] = acc[q] + bb;
  }
}

// ---------------------------------------------------------------------------
// Fused hot loop: hout[r,:] = elu( hin[r,:] @ W  +  x2[batch_idx[r], :] )
// 256 threads = 8 waves; block tile 64 rows x 128 cols.
// Wave w: row-tile rt = w>>1 (16 rows), col-group cg = w&1 (64 cols = 4 tiles).
// Split-bf16 WMMA: hi*hi + hi*lo + lo*hi with f32 accumulation (~fp32 accuracy).
// hi/lo planes are stored separately in EXACT fragment order, so A and B
// fragments are raw b128 loads with zero unpack ALU in the main loop.
// In-place safe: all reads of hin are staged to LDS before the barrier.
// ---------------------------------------------------------------------------
__global__ __launch_bounds__(256)
void k_fused(const float* hin, float* hout,
             const unsigned* __restrict__ WtHi, const unsigned* __restrict__ WtLo,
             const float* __restrict__ x2,
             const int* __restrict__ bidx, int n) {
  // 64 rows x 64 K-dwords (bf16 pairs), padded to 68 dwords: stride%64 = 4
  // -> consecutive rows start on distinct bank groups for ds_load_b128.
  __shared__ unsigned AsHi[64][68];
  __shared__ unsigned AsLo[64][68];
  const int tid = threadIdx.x;
  const int r0 = blockIdx.x * 64;

  // Stage A: 64x128 fp32 -> two bf16 planes in LDS (coalesced float4 loads).
#pragma unroll
  for (int it = 0; it < 8; ++it) {
    int e = tid + it * 256;      // float4 index 0..2047
    int rr = e >> 5;             // row in tile
    int c4 = e & 31;             // float4 column (K = 4*c4 .. 4*c4+3)
    int row = r0 + rr;
    float4 v;
    if (row < n) v = ((const float4*)hin)[(size_t)row * (D_ / 4) + c4];
    else { v.x = 0.f; v.y = 0.f; v.z = 0.f; v.w = 0.f; }
    unsigned hx, lx, hy, ly, hz, lz, hw, lw;
    split_bf16(v.x, hx, lx);
    split_bf16(v.y, hy, ly);
    split_bf16(v.z, hz, lz);
    split_bf16(v.w, hw, lw);
    AsHi[rr][2 * c4 + 0] = hx | (hy << 16);
    AsHi[rr][2 * c4 + 1] = hz | (hw << 16);
    AsLo[rr][2 * c4 + 0] = lx | (ly << 16);
    AsLo[rr][2 * c4 + 1] = lz | (lw << 16);
  }
  __syncthreads();

  const int lane = tid & 31;
  const int w = tid >> 5;
  const int rt = w >> 1;                 // 0..3
  const int cg = w & 1;                  // 0..1
  const int ln16 = lane & 15;
  const bool hiHalf = (lane >= 16);

  f32x8 acc[4];
#pragma unroll
  for (int ct = 0; ct < 4; ++ct)
#pragma unroll
    for (int i = 0; i < 8; ++i) acc[ct][i] = 0.f;

  const int arow = 16 * rt + ln16;
  // A 16-bit fragment layout: half-wave K runs kb..kb+7 and kb+16..kb+23,
  // kb = 0 (lanes 0-15) or 8 (lanes 16-31)  ->  dword offsets kd, kd+8.
  const int kdA = hiHalf ? 4 : 0;
  // B 16-bit fragment layout: same as A, K runs kb..kb+7 and kb+16..kb+23,
  // kb = 0 or 8 -> dword offsets kd, kd+8.
  const int kdB = hiHalf ? 4 : 0;

#pragma unroll
  for (int ks = 0; ks < 4; ++ks) {       // K = 128 in steps of 32 (16 dwords)
    // ---- A fragments: 4x ds_load_b128, no unpack ----
    Frag ahi, alo;
    const unsigned* ahp = &AsHi[arow][ks * 16 + kdA];
    const unsigned* alp = &AsLo[arow][ks * 16 + kdA];
    ahi.q[0] = *(const uint4*)(ahp);
    ahi.q[1] = *(const uint4*)(ahp + 8);
    alo.q[0] = *(const uint4*)(alp);
    alo.q[1] = *(const uint4*)(alp + 8);

#pragma unroll
    for (int ct = 0; ct < 4; ++ct) {
      int col = 64 * cg + 16 * ct + ln16;
      // ---- B fragments: 4x global_load_b128 from cache-resident planes ----
      Frag bhi, blo;
      const unsigned* bhp = WtHi + col * (D_ / 2) + ks * 16 + kdB;
      const unsigned* blp = WtLo + col * (D_ / 2) + ks * 16 + kdB;
      bhi.q[0] = *(const uint4*)(bhp);
      bhi.q[1] = *(const uint4*)(bhp + 8);
      blo.q[0] = *(const uint4*)(blp);
      blo.q[1] = *(const uint4*)(blp + 8);

      // D += A_hi*B_hi + A_hi*B_lo + A_lo*B_hi   (fp32-class accuracy)
      acc[ct] = __builtin_amdgcn_wmma_f32_16x16x32_bf16(
          false, ahi.v, false, bhi.v, (short)0, acc[ct], false, false);
      acc[ct] = __builtin_amdgcn_wmma_f32_16x16x32_bf16(
          false, ahi.v, false, blo.v, (short)0, acc[ct], false, false);
      acc[ct] = __builtin_amdgcn_wmma_f32_16x16x32_bf16(
          false, alo.v, false, bhi.v, (short)0, acc[ct], false, false);
    }
  }

  // Epilogue: + x2[batch_idx[row]], ELU, store.
  // C/D layout: VGPR i -> M = i (lanes 0-15) or i+8 (lanes 16-31); N = lane&15.
  const int rbase = r0 + 16 * rt + (hiHalf ? 8 : 0);
#pragma unroll
  for (int ct = 0; ct < 4; ++ct) {
    int col = 64 * cg + 16 * ct + ln16;
#pragma unroll
    for (int i = 0; i < 8; ++i) {
      int row = rbase + i;
      if (row < n) {
        int g = bidx[row];
        float x = acc[ct][i] + x2[(size_t)g * D_ + col];
        hout[(size_t)row * D_ + col] = (x > 0.f) ? x : (__expf(x) - 1.f);
      }
    }
  }
}

// ---------------------------------------------------------------------------
// fh[g][j] = relu( sum_k pooled[g][k]*Wf1[k][j] + bf1[j] ),  j < 256
// ---------------------------------------------------------------------------
__global__ __launch_bounds__(256) void k_head1(const float* __restrict__ pooled,
                                               const float* __restrict__ Wf1,
                                               const float* __restrict__ bf1,
                                               float* __restrict__ fh) {
  __shared__ float sp[8][D_];
  int g0 = blockIdx.x * 8;
  int tid = threadIdx.x;
  for (int it = 0; it < 4; ++it) {
    int e = tid + it * 256;
    int q = e >> 7;
    int k = e & 127;
    int g = g0 + q;
    sp[q][k] = (g < G_) ? pooled[(size_t)g * D_ + k] : 0.f;
  }
  __syncthreads();
  float acc[8];
#pragma unroll
  for (int q = 0; q < 8; ++q) acc[q] = 0.f;
  for (int k = 0; k < D_; ++k) {
    float w = Wf1[k * (2 * D_) + tid];
#pragma unroll
    for (int q = 0; q < 8; ++q) acc[q] += sp[q][k] * w;
  }
  float b = bf1[tid];
  for (int q = 0; q < 8; ++q) {
    int g = g0 + q;
    if (g < G_) fh[(size_t)g * (2 * D_) + tid] = fmaxf(acc[q] + b, 0.f);
  }
}

// ---------------------------------------------------------------------------
// out[g][t] = sum_j fh[g][j]*Wf2[j][t] + bf2[t]
// ---------------------------------------------------------------------------
__global__ void k_head2(const float* __restrict__ fh,
                        const float* __restrict__ Wf2,
                        const float* __restrict__ bf2,
                        float* __restrict__ out) {
  int i = blockIdx.x * blockDim.x + threadIdx.x;
  if (i >= G_ * T_) return;
  int g = i / T_;
  int t = i % T_;
  const float* fr = fh + (size_t)g * (2 * D_);
  float a = 0.f;
  for (int j = 0; j < 2 * D_; ++j) a += fr[j] * Wf2[j * T_ + t];
  out[i] = a + bf2[t];
}

extern "C" void kernel_launch(void* const* d_in, const int* in_sizes, int n_in,
                              void* d_out, int out_size, void* d_ws, size_t ws_size,
                              hipStream_t stream) {
  (void)in_sizes; (void)n_in; (void)out_size; (void)ws_size;

  const float* h_in = (const float*)d_in[0];
  const int* bidx = (const int*)d_in[1];
  const float* W[3]  = {(const float*)d_in[3], (const float*)d_in[7],  (const float*)d_in[11]};
  const float* b[3]  = {(const float*)d_in[4], (const float*)d_in[8],  (const float*)d_in[12]};
  const float* Ws[3] = {(const float*)d_in[5], (const float*)d_in[9],  (const float*)d_in[13]};
  const float* bs[3] = {(const float*)d_in[6], (const float*)d_in[10], (const float*)d_in[14]};
  const float* Wf1 = (const float*)d_in[15];
  const float* bf1 = (const float*)d_in[16];
  const float* Wf2 = (const float*)d_in[17];
  const float* bf2 = (const float*)d_in[18];
  float* out = (float*)d_out;

  // Workspace carve (needs ~277 MB: h ping buffer dominates).
  char* base = (char*)d_ws;
  size_t off = 0;
  auto carve = [&](size_t bytes) -> void* {
    void* r = base + off;
    off = (off + bytes + 255) & ~(size_t)255;
    return r;
  };
  float* hbuf   = (float*)carve((size_t)N_ * D_ * sizeof(float));
  float* pooled = (float*)carve((size_t)G_ * D_ * sizeof(float));
  float* x2     = (float*)carve((size_t)G_ * D_ * sizeof(float));
  float* fh     = (float*)carve((size_t)G_ * 2 * D_ * sizeof(float));
  unsigned* wHi[3];
  unsigned* wLo[3];
  for (int l = 0; l < 3; ++l) {
    wHi[l] = (unsigned*)carve((size_t)D_ * (D_ / 2) * sizeof(unsigned));
    wLo[l] = (unsigned*)carve((size_t)D_ * (D_ / 2) * sizeof(unsigned));
  }

  // Pre-split/transpose weights into separate bf16 hi/lo planes (tiny).
  for (int l = 0; l < 3; ++l)
    k_packW<<<(D_ * (D_ / 2) + 255) / 256, 256, 0, stream>>>(W[l], wHi[l], wLo[l]);

  const float* hprev = h_in;
  for (int l = 0; l < 3; ++l) {
    k_pool<<<G_, 128, 0, stream>>>(hprev, bidx, pooled, N_);
    k_x2<<<G_ / 8, 128, 0, stream>>>(pooled, Ws[l], bs[l], b[l], x2);
    k_fused<<<(N_ + 63) / 64, 256, 0, stream>>>(hprev, hbuf, wHi[l], wLo[l], x2, bidx, N_);
    hprev = hbuf;  // layers 2,3 run block-local in place on hbuf
  }

  k_pool<<<G_, 128, 0, stream>>>(hbuf, bidx, pooled, N_);
  k_head1<<<G_ / 8, 256, 0, stream>>>(pooled, Wf1, bf1, fh);
  k_head2<<<(G_ * T_ + 255) / 256, 256, 0, stream>>>(fh, Wf2, bf2, out);
}
